// DepthwiseSeparableDCNv2_6519760355535
// MI455X (gfx1250) — matrix operand order repaired and checked by reading, not compiled, not run on hardware
//
#include <hip/hip_runtime.h>
#include <hip/hip_bf16.h>

// ---------------------------------------------------------------------------
// DCNv2 depthwise-separable block for MI455X (gfx1250, wave32, WMMA).
//   1) zero halo + LDS-tiled transpose: x (NCHW f32) -> xh (NHWC bf16, padded
//      (H+2)x(W+2) with zero halo); weight repack
//   2) offset conv (3x3, 128->27) implicit GEMM, v_wmma_f32_16x16x32_bf16,
//      branch-free thanks to the halo (all A loads unconditional b128)
//   3) bilinear sampling + depthwise 3x3 (f32 gathers, LDS-shared offsets),
//      writes mid in NHWC bf16
//   4) pointwise 128->256 GEMM, v_wmma_f32_16x16x32_bf16
// NHWC staging makes every WMMA fragment a pair of 16-byte b128 loads
// (ISA A layout per lane = two contiguous 8-element K runs).
// ---------------------------------------------------------------------------

typedef __attribute__((ext_vector_type(16))) __bf16 v16bf;
typedef __attribute__((ext_vector_type(8)))  __bf16 v8bf;
typedef __attribute__((ext_vector_type(8)))  float  v8f;

constexpr int B_  = 4;
constexpr int C_  = 128;
constexpr int H_  = 128;
constexpr int W_  = 128;
constexpr int O_  = 256;
constexpr int HW_ = H_ * W_;
constexpr int HP_ = H_ + 2;                      // padded height
constexpr int WP_ = W_ + 2;                      // padded width
constexpr int NXH = B_ * HP_ * WP_ * C_;         // padded xh element count

// round-to-nearest-even f32 -> bf16 (stored as ushort)
__device__ __forceinline__ unsigned short to_bf16(float f) {
    unsigned u = __float_as_uint(f);
    unsigned r = u + 0x7FFFu + ((u >> 16) & 1u);
    return (unsigned short)(r >> 16);
}

// build a v16bf fragment from two contiguous 8-element runs
__device__ __forceinline__ v16bf frag_cat(v8bf lo, v8bf hi) {
    return __builtin_shufflevector(lo, hi, 0, 1, 2, 3, 4, 5, 6, 7,
                                   8, 9, 10, 11, 12, 13, 14, 15);
}

__device__ __forceinline__ v8bf ldg8(const __bf16* p) {
    return *(const v8bf*)p;
}

// ---------------------------------------------------------------------------
// zero the padded xh buffer (halo included); grid-stride, capture-safe
// ---------------------------------------------------------------------------
__global__ __launch_bounds__(256)
void k_zero(unsigned short* __restrict__ p, int n) {
    int i = blockIdx.x * 256 + threadIdx.x;
    const int stride = gridDim.x * 256;
    for (; i < n; i += stride) p[i] = 0;
}

// ---------------------------------------------------------------------------
// x (NCHW f32) -> xh (padded NHWC bf16) via LDS tile: coalesced on both sides.
// One block: all 128 channels x 16 consecutive pixels (same row, W%16==0).
// ---------------------------------------------------------------------------
__global__ __launch_bounds__(256)
void k_prep_x(const float* __restrict__ x, unsigned short* __restrict__ xh) {
    __shared__ float tile[C_][17];               // pad: stride 17 kills conflicts
    const int blk = blockIdx.x;                  // B * (HW/16) = 4096
    const int b   = blk >> 10;
    const int hw0 = (blk & 1023) << 4;
    const int h   = hw0 >> 7;                    // all 16 pixels share this row
    const int w0  = hw0 & 127;
    const int tid = threadIdx.x;

    for (int i = tid; i < C_ * 16; i += 256) {   // read: lanes walk pixels
        const int c = i >> 4, p = i & 15;
        tile[c][p] = x[((long)b * C_ + c) * HW_ + hw0 + p];
    }
    __syncthreads();
    for (int i = tid; i < 16 * C_; i += 256) {   // write: lanes walk channels
        const int p = i >> 7, c = i & 127;
        const long pix = ((long)b * HP_ + (h + 1)) * WP_ + (w0 + p + 1);
        xh[pix * C_ + c] = to_bf16(tile[c][p]);
    }
}

// wB2[tap][oc(0..31)][c] : oc padded 27->32 with zeros; w_off is (27,128,3,3).
// Lane-contiguous along c so offconv B-fragments are 32B vector loads.
__global__ __launch_bounds__(256)
void k_prep_w(const float* __restrict__ w_off, const float* __restrict__ w_pw,
              unsigned short* __restrict__ wB2, unsigned short* __restrict__ wpwh) {
    int i = blockIdx.x * 256 + threadIdx.x;
    if (i < 9 * 32 * 128) {
        const int c    = i & 127;
        const int rest = i >> 7;
        const int oc   = rest & 31;
        const int tap  = rest >> 5;
        wB2[i] = (oc < 27) ? to_bf16(w_off[(oc * 128 + c) * 9 + tap])
                           : (unsigned short)0;
    }
    if (i < O_ * C_) wpwh[i] = to_bf16(w_pw[i]);
}

// ---------------------------------------------------------------------------
// Offset conv: om[b, oc, h, w], oc in [0,27). One wave = 16 w-positions x 32 oc.
// K = 9 taps x 128 channels (chunks of 32). 72 WMMA / wave, branch-free,
// all unconditional b128 loads thanks to the zero halo.
// ---------------------------------------------------------------------------
__global__ __launch_bounds__(256)
void k_offconv(const unsigned short* __restrict__ xh,
               const unsigned short* __restrict__ wB2,
               const float* __restrict__ b_off,
               float* __restrict__ om) {
    const int lane = threadIdx.x & 31;
    const int wave = threadIdx.x >> 5;
    const int tile = blockIdx.x * 8 + wave;      // B*H*(W/16) = 4096 tiles
    const int b    = tile >> 10;
    const int rem  = tile & 1023;
    const int h    = rem >> 3;
    const int w0   = (rem & 7) << 4;
    const int m    = lane & 15;                  // A row / B column
    const int half = lane >> 4;                  // K half select

    const __bf16* xb = (const __bf16*)xh;
    const __bf16* wb = (const __bf16*)wB2;

    v8f acc0, acc1;
    const float bias0 = b_off[m];
    const float bias1 = (m < 11) ? b_off[16 + m] : 0.0f;
#pragma unroll
    for (int r = 0; r < 8; ++r) { acc0[r] = bias0; acc1[r] = bias1; }

#pragma unroll
    for (int tap = 0; tap < 9; ++tap) {
        const int ky = tap / 3, kx = tap % 3;    // halo: index with +0..2 directly
        // padded NHWC pixel base: (h-1+ky)+1 = h+ky ; (w-1+kx)+1 = w+kx
        const long pb = (((long)b * HP_ + (h + ky)) * WP_ + (w0 + m + kx)) * C_;
#pragma unroll
        for (int kc = 0; kc < 4; ++kc) {
            const int c0 = kc * 32;
            // A layout: K = 16*half + j (contiguous in c), matching B
            const v16bf A = frag_cat(ldg8(xb + pb + c0 + 16 * half),
                                     ldg8(xb + pb + c0 + 16 * half + 8));
            // B layout: lane = column, K = 16*half + j (contiguous in c)
            const __bf16* b0 = wb + ((long)(tap * 32 + m) * C_ + c0 + 16 * half);
            const __bf16* b1 = wb + ((long)(tap * 32 + 16 + m) * C_ + c0 + 16 * half);
            const v16bf Bf0 = frag_cat(ldg8(b0), ldg8(b0 + 8));
            const v16bf Bf1 = frag_cat(ldg8(b1), ldg8(b1 + 8));
            acc0 = __builtin_amdgcn_wmma_f32_16x16x32_bf16(false, A, false, Bf0,
                                                           (short)0, acc0, false, false);
            acc1 = __builtin_amdgcn_wmma_f32_16x16x32_bf16(false, A, false, Bf1,
                                                           (short)0, acc1, false, false);
        }
    }
#pragma unroll
    for (int r = 0; r < 8; ++r) {
        const int wq = w0 + r + 8 * half;        // ISA C/D layout: M = r + 8*half
        om[(((long)b * 27 + m) * H_ + h) * W_ + wq] = acc0[r];
        const int oc1 = 16 + m;
        if (oc1 < 27)
            om[(((long)b * 27 + oc1) * H_ + h) * W_ + wq] = acc1[r];
    }
}

// ---------------------------------------------------------------------------
// Bilinear sampling + depthwise 3x3. One block = (b, h, 32 w-positions).
// Offsets/mask shared across all 128 channels via LDS. x stays f32 (precision
// matters through the nonlinear gather); mid written NHWC bf16 for stage 4.
// ---------------------------------------------------------------------------
__global__ __launch_bounds__(256)
void k_sample(const float* __restrict__ x,
              const float* __restrict__ om,
              const float* __restrict__ w_dw,
              const float* __restrict__ b_dw,
              unsigned short* __restrict__ midh) {
    __shared__ int   s_idx[9][32][4];
    __shared__ float s_wgt[9][32][4];
    __shared__ float s_wdw[C_ * 9];
    __shared__ float s_bdw[C_];

    const int bi  = blockIdx.x;                  // B*H*(W/32) = 2048
    const int b   = bi >> 9;
    const int rem = bi & 511;
    const int h   = rem >> 2;
    const int w0  = (rem & 3) << 5;
    const int tid = threadIdx.x;

    for (int i = tid; i < C_ * 9; i += 256) s_wdw[i] = w_dw[i];
    for (int i = tid; i < C_;     i += 256) s_bdw[i] = b_dw[i];

    for (int i = tid; i < 9 * 32; i += 256) {
        const int tap = i >> 5;
        const int wp  = i & 31;
        const int w   = w0 + wp;
        const long ob = (long)b * 27 * HW_ + (long)h * W_ + w;
        const float dy = om[ob + (long)(2 * tap)     * HW_];
        const float dx = om[ob + (long)(2 * tap + 1) * HW_];
        float mk       = om[ob + (long)(18 + tap)    * HW_];
        mk = 1.0f / (1.0f + __expf(-mk));

        const float py = (float)h + (float)(tap / 3 - 1) + dy;
        const float px = (float)w + (float)(tap % 3 - 1) + dx;
        const float y0f = floorf(py), x0f = floorf(px);
        const int   iy0 = (int)y0f,   ix0 = (int)x0f;
        const float wy = py - y0f,    wx = px - x0f;
        const float wc[4] = {(1.0f - wy) * (1.0f - wx), (1.0f - wy) * wx,
                             wy * (1.0f - wx),          wy * wx};
        const int ys[4] = {iy0, iy0, iy0 + 1, iy0 + 1};
        const int xs[4] = {ix0, ix0 + 1, ix0, ix0 + 1};
#pragma unroll
        for (int c4 = 0; c4 < 4; ++c4) {
            const int yy = ys[c4], xx = xs[c4];
            const bool v = (yy >= 0) && (yy < H_) && (xx >= 0) && (xx < W_);
            const int yc = min(max(yy, 0), H_ - 1);
            const int xc = min(max(xx, 0), W_ - 1);
            s_idx[tap][wp][c4] = yc * W_ + xc;
            s_wgt[tap][wp][c4] = v ? wc[c4] * mk : 0.0f;
        }
    }
    __syncthreads();

    for (int it = tid; it < C_ * 32; it += 256) {
        const int c  = it >> 5;                  // gather-friendly: 32 lanes walk
        const int wp = it & 31;                  // neighboring pixels, one plane
        const float* xp = x + ((long)b * C_ + c) * HW_;
        float acc = 0.0f;
#pragma unroll
        for (int tap = 0; tap < 9; ++tap) {
            const float v = s_wgt[tap][wp][0] * xp[s_idx[tap][wp][0]]
                          + s_wgt[tap][wp][1] * xp[s_idx[tap][wp][1]]
                          + s_wgt[tap][wp][2] * xp[s_idx[tap][wp][2]]
                          + s_wgt[tap][wp][3] * xp[s_idx[tap][wp][3]];
            acc += v * s_wdw[c * 9 + tap];
        }
        acc += s_bdw[c];
        // NHWC store for stage 4 (1/36 of this kernel's traffic)
        midh[((long)b * HW_ + (long)h * W_ + w0 + wp) * C_ + c] = to_bf16(acc);
    }
}

// ---------------------------------------------------------------------------
// Pointwise 128 -> 256 GEMM. One wave = 16 pixels x 16 out-ch, loops all 16
// o-tiles reusing the A fragments (K=128 preloaded). 64 WMMA / wave.
// ---------------------------------------------------------------------------
__global__ __launch_bounds__(256)
void k_pointwise(const unsigned short* __restrict__ midh,
                 const unsigned short* __restrict__ wpwh,
                 const float* __restrict__ b_pw,
                 float* __restrict__ out) {
    const int lane = threadIdx.x & 31;
    const int wave = threadIdx.x >> 5;
    const int tile = blockIdx.x * 8 + wave;      // B*HW/16 = 4096 tiles
    const int b    = tile >> 10;
    const int s0   = (tile & 1023) << 4;
    const int m    = lane & 15;
    const int half = lane >> 4;

    const __bf16* mb = (const __bf16*)midh;
    const __bf16* wb = (const __bf16*)wpwh;

    v16bf A[4];
    const long abase = ((long)b * HW_ + s0 + m) * C_;   // NHWC pixel base
#pragma unroll
    for (int kc = 0; kc < 4; ++kc) {
        const __bf16* p = mb + abase + kc * 32 + 16 * half;
        A[kc] = frag_cat(ldg8(p), ldg8(p + 8));
    }

    for (int ot = 0; ot < 16; ++ot) {
        const int o = ot * 16 + m;
        const float bias = b_pw[o];
        v8f acc;
#pragma unroll
        for (int r = 0; r < 8; ++r) acc[r] = bias;
#pragma unroll
        for (int kc = 0; kc < 4; ++kc) {
            const __bf16* p = wb + (long)o * C_ + kc * 32 + 16 * half;
            const v16bf Bf = frag_cat(ldg8(p), ldg8(p + 8));
            acc = __builtin_amdgcn_wmma_f32_16x16x32_bf16(false, A[kc], false, Bf,
                                                          (short)0, acc, false, false);
        }
#pragma unroll
        for (int r = 0; r < 8; ++r)
            out[((long)b * O_ + o) * HW_ + s0 + r + 8 * half] = acc[r];
    }
}

// ---------------------------------------------------------------------------
// launch
// ---------------------------------------------------------------------------
extern "C" void kernel_launch(void* const* d_in, const int* in_sizes, int n_in,
                              void* d_out, int out_size, void* d_ws, size_t ws_size,
                              hipStream_t stream) {
    const float* x     = (const float*)d_in[0];
    const float* w_off = (const float*)d_in[1];
    const float* b_off = (const float*)d_in[2];
    const float* w_dw  = (const float*)d_in[3];
    const float* b_dw  = (const float*)d_in[4];
    const float* w_pw  = (const float*)d_in[5];
    const float* b_pw  = (const float*)d_in[6];
    float* out = (float*)d_out;

    char* ws = (char*)d_ws;
    // workspace layout (bytes), all 16B aligned
    unsigned short* xh   = (unsigned short*)(ws + 0);          // 17,305,600 (padded)
    unsigned short* wB2  = (unsigned short*)(ws + 17305600);   //     73,728
    unsigned short* wpwh = (unsigned short*)(ws + 17379328);   //     65,536
    float*          om   = (float*)        (ws + 17444864);    //  7,077,888
    unsigned short* midh = (unsigned short*)(ws + 24522752);   // 16,777,216
    (void)in_sizes; (void)n_in; (void)out_size; (void)ws_size;

    k_zero     <<<8192, 256, 0, stream>>>((unsigned short*)xh, NXH);
    k_prep_x   <<<4096, 256, 0, stream>>>(x, xh);
    k_prep_w   <<<144,  256, 0, stream>>>(w_off, w_pw, wB2, wpwh);
    k_offconv  <<<512,  256, 0, stream>>>(xh, wB2, b_off, om);
    k_sample   <<<2048, 256, 0, stream>>>(x, om, w_dw, b_dw, midh);
    k_pointwise<<<512,  256, 0, stream>>>(midh, wpwh, b_pw, out);
}
